// Attention_56435870269974
// MI455X (gfx1250) — matrix operand order repaired and checked by reading, not compiled, not run on hardware
//
#include <hip/hip_runtime.h>
#include <cstdint>

typedef _Float16 h16;
typedef __attribute__((ext_vector_type(4)))  _Float16 v4h;
typedef __attribute__((ext_vector_type(8)))  _Float16 v8h;
typedef __attribute__((ext_vector_type(16))) _Float16 v16h;
typedef __attribute__((ext_vector_type(8)))  float    v8f;

#define SEQ 2048
#define DIM 4096
#define NH  32
#define HD  128

// ---------- WMMA fragment helpers (wave32 layouts, CDNA5 ISA 7.12.2) ----------
// A-matrix 16x32 f16: lane(0-15)=row M, holds K = koff..koff+7 and koff+16..koff+23
// where koff = (lane>>4)*8.
__device__ __forceinline__ v16h ldfragA(const h16* p) {
  v8h lo = *(const v8h*)(p);
  v8h hi = *(const v8h*)(p + 16);
  return __builtin_shufflevector(lo, hi, 0,1,2,3,4,5,6,7,8,9,10,11,12,13,14,15);
}
// B-matrix 32x16 f16 (loaded from B^T row-major [N][K]): lane(0-15)=col N,
// holds K = koff..koff+7 and koff+16..koff+23 where koff = (lane>>4)*8.
__device__ __forceinline__ v16h ldfragB(const h16* p) {
  v8h lo = *(const v8h*)(p);
  v8h hi = *(const v8h*)(p + 16);
  return __builtin_shufflevector(lo, hi, 0,1,2,3,4,5,6,7,8,9,10,11,12,13,14,15);
}
__device__ __forceinline__ v8f wmma16(v16h a, v16h b, v8f c) {
  return __builtin_amdgcn_wmma_f32_16x16x32_f16(false, a, false, b, (short)0, c,
                                                false, false);
}

// ---------- elementwise f32 -> f16 convert (4 elems/thread) ----------
__global__ void cvt_f16_kernel(const float* __restrict__ in,
                               h16* __restrict__ out, int n4) {
  int t = blockIdx.x * blockDim.x + threadIdx.x;
  if (t >= n4) return;
  float4 v = ((const float4*)in)[t];
  v4h o;
  o[0] = (h16)v.x; o[1] = (h16)v.y; o[2] = (h16)v.z; o[3] = (h16)v.w;
  ((v4h*)out)[t] = o;
}

// ---------- tiled transpose + convert: in f32 [R][C] -> out f16 [C][R] ----------
__global__ __launch_bounds__(256) void transpose_cvt_kernel(
    const float* __restrict__ in, h16* __restrict__ out, int R, int C) {
  __shared__ float tile[32][33];
  int tx = threadIdx.x & 31;
  int ty = threadIdx.x >> 5;              // 0..7
  int c0 = blockIdx.x * 32;
  int r0 = blockIdx.y * 32;
#pragma unroll
  for (int j = 0; j < 32; j += 8)
    tile[ty + j][tx] = in[(size_t)(r0 + ty + j) * C + (c0 + tx)];
  __syncthreads();
#pragma unroll
  for (int j = 0; j < 32; j += 8)
    out[(size_t)(c0 + ty + j) * R + (r0 + tx)] = (h16)tile[tx][ty + j];
}

// ---------- GEMM: C[M][N] f32 = A[M][K] f16 x (BT[N][K] f16)^T ----------
// One wave owns a 64x64 C macro-tile: 16 accumulator C-tiles (128 VGPRs),
// 16 v_wmma_f32_16x16x32_f16 per 32-wide k-step.
__global__ __launch_bounds__(256) void gemm_f16_nt_kernel(
    const h16* __restrict__ A, const h16* __restrict__ BT,
    float* __restrict__ C, int M, int N, int K) {
  const int lane = threadIdx.x & 31;
  const int wib  = threadIdx.x >> 5;
  const int tilesM = M >> 6, tilesN = N >> 6;
  const int wtile = blockIdx.x * 8 + wib;
  if (wtile >= tilesM * tilesN) return;    // wave-uniform
  const int tm = wtile % tilesM, tn = wtile / tilesM;
  const int m0 = tm << 6, n0 = tn << 6;

  const int rA    = lane & 15;
  const int koffA = (lane >> 4) * 8;
  const int koffB = (lane >> 4) * 8;

  v8f zero8 = {0.f,0.f,0.f,0.f,0.f,0.f,0.f,0.f};
  v8f acc[4][4];
#pragma unroll
  for (int i = 0; i < 4; i++)
#pragma unroll
    for (int j = 0; j < 4; j++) acc[i][j] = zero8;

  for (int k0 = 0; k0 < K; k0 += 32) {
    if (k0 + 32 < K) {   // hint next k-panel into cache: global_prefetch_b8
      __builtin_prefetch(A  + (size_t)(m0 + rA) * K + k0 + 32 + koffA, 0, 1);
      __builtin_prefetch(BT + (size_t)(n0 + rA) * K + k0 + 32 + koffB, 0, 1);
    }
    v16h a[4], b[4];
#pragma unroll
    for (int i = 0; i < 4; i++)
      a[i] = ldfragA(A + (size_t)(m0 + i * 16 + rA) * K + k0 + koffA);
#pragma unroll
    for (int i = 0; i < 4; i++)
      b[i] = ldfragB(BT + (size_t)(n0 + i * 16 + rA) * K + k0 + koffB);
#pragma unroll
    for (int mi = 0; mi < 4; mi++)
#pragma unroll
      for (int ni = 0; ni < 4; ni++)
        acc[mi][ni] = wmma16(a[mi], b[ni], acc[mi][ni]);
  }
  // C-layout: element (m = v + 8*(lane>>4), n = lane&15) in VGPR v
  const int ms = (lane >> 4) * 8, nc = lane & 15;
#pragma unroll
  for (int mi = 0; mi < 4; mi++)
#pragma unroll
    for (int ni = 0; ni < 4; ni++) {
      float* cp = C + (size_t)(m0 + mi * 16 + ms) * N + (n0 + ni * 16 + nc);
#pragma unroll
      for (int v = 0; v < 8; v++) cp[(size_t)v * N] = acc[mi][ni][v];
    }
}

// ---------- RoPE + repack Q,K: f32 [s][4096] -> f16 [h][s][128] ----------
__global__ void rope_pack_kernel(const float* __restrict__ Qf,
                                 const float* __restrict__ Kf,
                                 const float* __restrict__ fr,
                                 const float* __restrict__ fi,
                                 h16* __restrict__ Qh, h16* __restrict__ Kh) {
  int t = blockIdx.x * blockDim.x + threadIdx.x;
  if (t >= SEQ * NH * (HD / 2)) return;
  int i = t & 63;            // pair index 0..63
  int h = (t >> 6) & 31;
  int s = t >> 11;
  float cr = fr[s * 64 + i], ci = fi[s * 64 + i];
  size_t src = (size_t)s * DIM + h * HD + 2 * i;
  size_t dst = ((size_t)h * SEQ + s) * HD + 2 * i;
  float qa = Qf[src], qb = Qf[src + 1];
  Qh[dst]     = (h16)(qa * cr - qb * ci);
  Qh[dst + 1] = (h16)(qa * ci + qb * cr);
  float ka = Kf[src], kb = Kf[src + 1];
  Kh[dst]     = (h16)(ka * cr - kb * ci);
  Kh[dst + 1] = (h16)(ka * ci + kb * cr);
}

// ---------- Flash attention: 1 wave = (head, 16-row query block) ----------
// Qh,Kh: [h][s][128] f16 ; Vt: [h][d][s] f16 ; Oh: [s][4096] f16
__global__ __launch_bounds__(128) void attn_kernel(
    const h16* __restrict__ Qh, const h16* __restrict__ Kh,
    const h16* __restrict__ Vt, h16* __restrict__ Oh) {
  __shared__ h16 plds[4][16 * 40];   // 80B row pitch -> every v8h reload 16B-aligned
  const int lane = threadIdx.x & 31;
  const int w    = threadIdx.x >> 5;
  const int wid  = blockIdx.x * 4 + w;
  const int head = wid >> 7;          // 128 qblocks / head
  const int qb   = wid & 127;
  const int q0   = qb * 16;

  const h16* Q  = Qh + (size_t)head * SEQ * HD;
  const h16* Kp = Kh + (size_t)head * SEQ * HD;
  const h16* Vp = Vt + (size_t)head * HD * SEQ;

  const int rA = lane & 15, koffA = (lane >> 4) * 8, koffB = (lane >> 4) * 8;
  const int ms = (lane >> 4) * 8, nc = lane & 15;
  const float scale = 0.08838834764831845f;   // 1/sqrt(128)

  v16h qf[4];
#pragma unroll
  for (int c = 0; c < 4; c++)
    qf[c] = ldfragA(Q + (size_t)(q0 + rA) * HD + c * 32 + koffA);

  v8f zero8 = {0.f,0.f,0.f,0.f,0.f,0.f,0.f,0.f};
  v8f o[8];
#pragma unroll
  for (int n = 0; n < 8; n++) o[n] = zero8;
  float mrow[8], lrow[8];
#pragma unroll
  for (int v = 0; v < 8; v++) { mrow[v] = -3.0e38f; lrow[v] = 0.f; }

  h16* myp = &plds[w][0];

  for (int kb0 = 0; kb0 <= q0 + 15; kb0 += 32) {
    // ---- scores: two 16x16 C-tiles over key block [kb0, kb0+32) ----
    v8f s0 = zero8, s1 = zero8;
#pragma unroll
    for (int c = 0; c < 4; c++) {
      v16h kfr = ldfragB(Kp + (size_t)(kb0 + rA) * HD + c * 32 + koffB);
      s0 = wmma16(qf[c], kfr, s0);
    }
#pragma unroll
    for (int c = 0; c < 4; c++) {
      v16h kfr = ldfragB(Kp + (size_t)(kb0 + 16 + rA) * HD + c * 32 + koffB);
      s1 = wmma16(qf[c], kfr, s1);
    }
    // ---- causal mask + online softmax (per C-layout row v+ms) ----
    float pr0[8], pr1[8], corr[8];
#pragma unroll
    for (int v = 0; v < 8; v++) {
      int qi = q0 + v + ms;
      float a0 = s0[v] * scale + ((kb0 + nc)      > qi ? -1.0e9f : 0.f);
      float a1 = s1[v] * scale + ((kb0 + 16 + nc) > qi ? -1.0e9f : 0.f);
      float mx = fmaxf(a0, a1);
#pragma unroll
      for (int off = 1; off < 16; off <<= 1)      // stays inside 16-lane half
        mx = fmaxf(mx, __shfl_xor(mx, off, 32));
      float nm = fmaxf(mrow[v], mx);
      float p0 = __expf(a0 - nm), p1 = __expf(a1 - nm);
      float rs = p0 + p1;
#pragma unroll
      for (int off = 1; off < 16; off <<= 1)
        rs += __shfl_xor(rs, off, 32);
      corr[v] = __expf(mrow[v] - nm);
      mrow[v] = nm;
      lrow[v] = lrow[v] * corr[v] + rs;
      pr0[v] = p0; pr1[v] = p1;
    }
#pragma unroll
    for (int n = 0; n < 8; n++)
#pragma unroll
      for (int v = 0; v < 8; v++) o[n][v] *= corr[v];
    // ---- P (C-layout) -> LDS -> A-layout fragment (same-wave DS is in-order) ----
#pragma unroll
    for (int v = 0; v < 8; v++) {
      int row = v + ms;
      myp[row * 40 + nc]      = (h16)pr0[v];
      myp[row * 40 + 16 + nc] = (h16)pr1[v];
    }
    v16h pa = ldfragA(myp + rA * 40 + koffA);
    // ---- PV: 8 n-tiles of V (B^T rows = head-dim rows, contiguous along s) ----
#pragma unroll
    for (int n = 0; n < 8; n++) {
      v16h vf = ldfragB(Vp + (size_t)(n * 16 + rA) * SEQ + kb0 + koffB);
      o[n] = wmma16(pa, vf, o[n]);
    }
  }
  // ---- normalize and store f16 attention output [s][h*128+d] ----
#pragma unroll
  for (int v = 0; v < 8; v++) lrow[v] = 1.0f / lrow[v];
#pragma unroll
  for (int n = 0; n < 8; n++)
#pragma unroll
    for (int v = 0; v < 8; v++) {
      size_t oi = (size_t)(q0 + v + ms) * DIM + head * HD + n * 16 + nc;
      Oh[oi] = (h16)(o[n][v] * lrow[v]);
    }
}

// ---------------------------- host launcher ----------------------------
extern "C" void kernel_launch(void* const* d_in, const int* in_sizes, int n_in,
                              void* d_out, int out_size, void* d_ws, size_t ws_size,
                              hipStream_t stream) {
  const float* x  = (const float*)d_in[0];
  const float* wq = (const float*)d_in[1];
  const float* wk = (const float*)d_in[2];
  const float* wv = (const float*)d_in[3];
  const float* wo = (const float*)d_in[4];
  const float* fr = (const float*)d_in[7];
  const float* fi = (const float*)d_in[8];
  float* out = (float*)d_out;

  char* ws = (char*)d_ws;
  const size_t SZ_XH = (size_t)SEQ * DIM * 2;   // 16 MB
  const size_t SZ_WT = (size_t)DIM * DIM * 2;   // 32 MB
  const size_t SZ_F  = (size_t)SEQ * DIM * 4;   // 32 MB
  const size_t SZ_H  = (size_t)SEQ * DIM * 2;   // 16 MB
  h16*   xh  = (h16*)(ws);
  h16*   wqT = (h16*)(ws + SZ_XH);
  h16*   wkT = (h16*)(ws + SZ_XH + SZ_WT);
  h16*   wvT = (h16*)(ws + SZ_XH + 2 * SZ_WT);
  h16*   woT = (h16*)(ws + SZ_XH + 3 * SZ_WT);
  float* Qf  = (float*)(ws + SZ_XH + 4 * SZ_WT);
  float* Kf  = (float*)(ws + SZ_XH + 4 * SZ_WT + SZ_F);
  float* Vf  = (float*)(ws + SZ_XH + 4 * SZ_WT + 2 * SZ_F);
  h16*   Qh  = (h16*)(ws + SZ_XH + 4 * SZ_WT + 3 * SZ_F);
  h16*   Kh  = (h16*)(ws + SZ_XH + 4 * SZ_WT + 3 * SZ_F + SZ_H);
  h16*   Vt  = (h16*)(ws + SZ_XH + 4 * SZ_WT + 3 * SZ_F + 2 * SZ_H);
  h16*   Ah  = (h16*)(ws + SZ_XH + 4 * SZ_WT + 3 * SZ_F + 3 * SZ_H);

  // 1) x -> f16
  {
    int n4 = SEQ * DIM / 4;
    cvt_f16_kernel<<<(n4 + 255) / 256, 256, 0, stream>>>(x, xh, n4);
  }
  // 2) weight transposes -> f16 [N][K]
  {
    dim3 g(DIM / 32, DIM / 32);
    transpose_cvt_kernel<<<g, 256, 0, stream>>>(wq, wqT, DIM, DIM);
    transpose_cvt_kernel<<<g, 256, 0, stream>>>(wk, wkT, DIM, DIM);
    transpose_cvt_kernel<<<g, 256, 0, stream>>>(wv, wvT, DIM, DIM);
    transpose_cvt_kernel<<<g, 256, 0, stream>>>(wo, woT, DIM, DIM);
  }
  // 3) QKV projections (f32 accum)
  {
    int blocks = (SEQ / 64) * (DIM / 64) / 8;   // 256
    gemm_f16_nt_kernel<<<blocks, 256, 0, stream>>>(xh, wqT, Qf, SEQ, DIM, DIM);
    gemm_f16_nt_kernel<<<blocks, 256, 0, stream>>>(xh, wkT, Kf, SEQ, DIM, DIM);
    gemm_f16_nt_kernel<<<blocks, 256, 0, stream>>>(xh, wvT, Vf, SEQ, DIM, DIM);
  }
  // 4) RoPE(Q,K) -> per-head f16 ; V -> [h][d][s] f16 (transpose)
  {
    int total = SEQ * NH * (HD / 2);
    rope_pack_kernel<<<(total + 255) / 256, 256, 0, stream>>>(Qf, Kf, fr, fi, Qh, Kh);
    dim3 g(DIM / 32, SEQ / 32);
    transpose_cvt_kernel<<<g, 256, 0, stream>>>(Vf, Vt, SEQ, DIM);
  }
  // 5) flash attention (32 heads x 128 qblocks, 4 waves/block)
  attn_kernel<<<(NH * 128) / 4, 128, 0, stream>>>(Qh, Kh, Vt, Ah);
  // 6) output projection -> d_out (f32)
  {
    int blocks = (SEQ / 64) * (DIM / 64) / 8;
    gemm_f16_nt_kernel<<<blocks, 256, 0, stream>>>(Ah, woT, out, SEQ, DIM, DIM);
  }
}
